// AttentionDecoder_19361712571451
// MI455X (gfx1250) — matrix operand order repaired and checked by reading, not compiled, not run on hardware
//
#include <hip/hip_runtime.h>
#include <hip/hip_bf16.h>
#include <math.h>

// ---------------------------------------------------------------------------
// AttentionDecoder for MI455X (gfx1250, wave32, WMMA bf16, async LDS DMA)
// E=256 H=8 DH=32 N=200000
//
// K_w : Wk,Wv f32 -> bf16 in ws
// K0  : query = ctx@Wc.T+bc ; q = query@wq.T+bq ; G[h][i]=sum_d q[h,d]*wk2[h,d][i]
// K1  : per 64-row tile: keys_raw = X@Wk.T+bk (WMMA bf16, async double-buffered
//       weight slices), row-normalize, store 1/||.||, logits = keys.G/sqrt(32)
// K2a/b: per-head online-softmax (max, denom) over N
// K3  : values = normalize(X@Wv.T+bv) (WMMA), Z[h][i] += w[h][n]*values[n][i]
// K4  : attn = Z@wv2 + bv2 -> out_proj -> Wo -> normalize -> a ; u = Wk^T a
// K5  : scores_n = valid ? (x_n.u + c0)*invnorm_n : NEG
// K6a/b + K7 : softmax(scores) -> probs (d_out)
// ---------------------------------------------------------------------------

#define EDIM   256
#define NNODES 200000
#define NEGV   (-1000000000.0f)

typedef __attribute__((ext_vector_type(16))) __bf16 v16bf;
typedef __attribute__((ext_vector_type(8)))  __bf16 v8bf;
typedef __attribute__((ext_vector_type(8)))  float  v8f;

// workspace byte offsets
#define OFF_WK      0u          // bf16 [65536]
#define OFF_WV      131072u     // bf16 [65536]
#define OFF_G       262144u     // f32  [2048]
#define OFF_SCALE   270336u     // f32  [N]     (1/||keys_raw||)
#define OFF_LOGITS  1070336u    // f32  [8*N]
#define OFF_SCORES  7470336u    // f32  [N]
#define OFF_PART    8270336u    // f32  [1024]
#define OFF_HMAX    8274432u    // f32  [8]
#define OFF_HINV    8274496u    // f32  [8]
#define OFF_Z       8274560u    // f32  [2048]
#define OFF_U       8282752u    // f32  [256]
#define OFF_C0      8283776u    // f32  [1]
#define OFF_SPART   8283904u    // f32  [128]
#define OFF_SMAX    8284416u    // f32  [1]
#define OFF_SINV    8284480u    // f32  [1]

// X tile in LDS: 64 rows, pitch 264 bf16 (528B = 33*16B -> aligned, banks spread)
#define XPITCH 264
// weight slice in LDS: 256 rows x 32 cols, pitch 40 bf16 (80B = 5*16B)
#define WPITCH 40
#define WSLICE (256 * WPITCH)   // elements per slice buffer

__device__ __forceinline__ v16bf load_a_frag(const __bf16* sX, int m0, int lane, int kt) {
    const int l15 = lane & 15, hi = lane >> 4;
    const __bf16* p = sX + (m0 + l15) * XPITCH + kt * 32 + 8 * hi;
    v8bf lo = *(const v8bf*)p;
    v8bf hh = *(const v8bf*)(p + 16);
    v16bf a;
#pragma unroll
    for (int e = 0; e < 8; ++e) { a[e] = lo[e]; a[e + 8] = hh[e]; }
    return a;
}

__device__ __forceinline__ v16bf load_b_frag(const __bf16* sW, int jt, int lane) {
    const int l15 = lane & 15, hi = lane >> 4;
    const __bf16* p = sW + (jt * 16 + l15) * WPITCH + 8 * hi;
    v8bf b0 = *(const v8bf*)p;
    v8bf b1 = *(const v8bf*)(p + 16);
    v16bf b;
#pragma unroll
    for (int e = 0; e < 8; ++e) { b[e] = b0[e]; b[e + 8] = b1[e]; }
    return b;
}

// Async global->LDS copy of one 256x32 bf16 weight slice (16KB) as 1024 x 16B
// chunks. GLOBAL_LOAD_ASYNC_TO_LDS_B128, GVS mode, tracked with ASYNCcnt.
__device__ __forceinline__ void stage_slice_async(const __bf16* wsrc, int kt,
                                                  __bf16* dstLds, int tid) {
    for (int i = tid; i < 1024; i += 128) {
        int r = i >> 2, c = i & 3;                       // 4 x 16B per 64B row chunk
        unsigned voff = (unsigned)(r * 512 + kt * 64 + c * 16);   // src byte offset
        unsigned lds  = (unsigned)(size_t)(dstLds + r * WPITCH + c * 8);
        asm volatile("global_load_async_to_lds_b128 %0, %1, %2"
                     :: "v"(lds), "v"(voff), "s"(wsrc) : "memory");
    }
}

__device__ __forceinline__ void wait_async_all() {
    asm volatile("s_wait_asynccnt 0" ::: "memory");
}

// -------------------------------- K_w --------------------------------------
__global__ __launch_bounds__(256) void kw_conv(const float* __restrict__ Wk,
                                               const float* __restrict__ Wv,
                                               __bf16* __restrict__ wkbf,
                                               __bf16* __restrict__ wvbf) {
    int i = blockIdx.x * 256 + threadIdx.x;
    if (i < EDIM * EDIM) { wkbf[i] = (__bf16)Wk[i]; wvbf[i] = (__bf16)Wv[i]; }
}

// -------------------------------- K0 ---------------------------------------
__global__ __launch_bounds__(256) void k0_query(const float* __restrict__ ctx,
                                                const float* __restrict__ Wc,
                                                const float* __restrict__ bc,
                                                const float* __restrict__ in_w,
                                                const float* __restrict__ in_b,
                                                float* __restrict__ G) {
    __shared__ float sQuery[256];
    __shared__ float sQ[256];
    int t = threadIdx.x;
    float qv = bc[t];
    for (int j = 0; j < 768; ++j) qv += ctx[j] * Wc[(size_t)t * 768 + j];
    sQuery[t] = qv;
    __syncthreads();
    float q2 = in_b[t];
    for (int j = 0; j < 256; ++j) q2 += in_w[(size_t)t * 256 + j] * sQuery[j];
    sQ[t] = q2;
    __syncthreads();
    for (int h = 0; h < 8; ++h) {
        float g = 0.f;
        for (int d = 0; d < 32; ++d)
            g += sQ[h * 32 + d] * in_w[(size_t)(256 + h * 32 + d) * 256 + t];
        G[h * 256 + t] = g;
    }
}

// -------------------------------- K1 ---------------------------------------
__global__ __launch_bounds__(128) void k1_keys_logits(
        const float* __restrict__ node, const unsigned char* __restrict__ mask,
        const __bf16* __restrict__ wkbf, const float* __restrict__ bk,
        const float* __restrict__ G, float* __restrict__ scaleOut,
        float* __restrict__ logitsOut) {
    __shared__ __bf16 sX[64 * XPITCH];      // 33792 B
    __shared__ __bf16 sW[2 * WSLICE];       // 40960 B (double-buffered slices)
    __shared__ float  sG[2048];             //  8192 B
    const int tid = threadIdx.x;
    const int lane = tid & 31, wv = tid >> 5;
    const int l15 = lane & 15, hi = lane >> 4;
    const int base = blockIdx.x * 64;
    const int m0 = wv * 16;

    // prologue: kick off async DMA of weight slice 0 while staging X in bf16
    stage_slice_async(wkbf, 0, sW, tid);

    const float4* nf4 = (const float4*)node;
    for (int i = tid; i < 64 * 64; i += 128) {
        int r = i >> 6, c4 = i & 63;
        int n = base + r;
        float4 v = make_float4(0.f, 0.f, 0.f, 0.f);
        if (n < NNODES) v = nf4[(size_t)n * 64 + c4];
        __bf16* d = &sX[r * XPITCH + c4 * 4];
        d[0] = (__bf16)v.x; d[1] = (__bf16)v.y; d[2] = (__bf16)v.z; d[3] = (__bf16)v.w;
    }
    for (int i = tid; i < 2048; i += 128) sG[i] = G[i];

    v8f acc[16];
#pragma unroll
    for (int jt = 0; jt < 16; ++jt)
#pragma unroll
        for (int e = 0; e < 8; ++e) acc[jt][e] = 0.f;

    wait_async_all();
    __syncthreads();

    for (int kt = 0; kt < 8; ++kt) {
        const __bf16* cur = sW + (kt & 1) * WSLICE;
        if (kt < 7) stage_slice_async(wkbf, kt + 1, sW + ((kt + 1) & 1) * WSLICE, tid);

        v16bf A = load_a_frag(sX, m0, lane, kt);
        v16bf Bc = load_b_frag(cur, 0, lane);
#pragma unroll
        for (int jt = 0; jt < 16; ++jt) {
            v16bf Bn = Bc;
            if (jt < 15) Bn = load_b_frag(cur, jt + 1, lane);
            acc[jt] = __builtin_amdgcn_wmma_f32_16x16x32_bf16(
                false, A, false, Bc, (short)0, acc[jt], false, false);
            Bc = Bn;
        }
        wait_async_all();
        __syncthreads();
    }

    // bias + row norms
    float ss[8];
#pragma unroll
    for (int r = 0; r < 8; ++r) ss[r] = 0.f;
#pragma unroll
    for (int jt = 0; jt < 16; ++jt) {
        float bkv = bk[jt * 16 + l15];
#pragma unroll
        for (int r = 0; r < 8; ++r) {
            float t = acc[jt][r] + bkv;
            acc[jt][r] = t;
            ss[r] += t * t;
        }
    }
#pragma unroll
    for (int m = 1; m <= 8; m <<= 1)
#pragma unroll
        for (int r = 0; r < 8; ++r) ss[r] += __shfl_xor(ss[r], m, 32);
    float sc[8];
#pragma unroll
    for (int r = 0; r < 8; ++r) sc[r] = 1.f / fmaxf(sqrtf(ss[r]), 1e-12f);
#pragma unroll
    for (int jt = 0; jt < 16; ++jt)
#pragma unroll
        for (int r = 0; r < 8; ++r) acc[jt][r] *= sc[r];
    if (l15 == 0) {
#pragma unroll
        for (int r = 0; r < 8; ++r) {
            int n = base + m0 + hi * 8 + r;
            if (n < NNODES) scaleOut[n] = sc[r];
        }
    }
    // logits: keys . G[h]
    float lg[8][8];
#pragma unroll
    for (int r = 0; r < 8; ++r)
#pragma unroll
        for (int h = 0; h < 8; ++h) lg[r][h] = 0.f;
#pragma unroll
    for (int jt = 0; jt < 16; ++jt) {
        int c = jt * 16 + l15;
#pragma unroll
        for (int h = 0; h < 8; ++h) {
            float g = sG[h * 256 + c];
#pragma unroll
            for (int r = 0; r < 8; ++r) lg[r][h] += g * acc[jt][r];
        }
    }
#pragma unroll
    for (int m = 1; m <= 8; m <<= 1)
#pragma unroll
        for (int r = 0; r < 8; ++r)
#pragma unroll
            for (int h = 0; h < 8; ++h) lg[r][h] += __shfl_xor(lg[r][h], m, 32);
    if (l15 == 0) {
#pragma unroll
        for (int r = 0; r < 8; ++r) {
            int n = base + m0 + hi * 8 + r;
            if (n < NNODES) {
                bool inval = (mask[n] != 0);
#pragma unroll
                for (int h = 0; h < 8; ++h)
                    logitsOut[(size_t)h * NNODES + n] =
                        inval ? NEGV : lg[r][h] * 0.17677669529663689f; // 1/sqrt(32)
            }
        }
    }
}

// -------------------------------- K2 ---------------------------------------
__device__ __forceinline__ void ms_combine(float& m, float& s, float m2, float s2) {
    float nm = fmaxf(m, m2);
    s = s * __expf(m - nm) + s2 * __expf(m2 - nm);
    m = nm;
}

__global__ __launch_bounds__(256) void k2a_head_partials(const float* __restrict__ logits,
                                                         float* __restrict__ part) {
    int h = blockIdx.x >> 6, chunk = blockIdx.x & 63;
    int start = chunk * 3125, end = min(start + 3125, NNODES);
    float m = -3.0e38f, s = 0.f;
    for (int i = start + threadIdx.x; i < end; i += 256) {
        float x = logits[(size_t)h * NNODES + i];
        ms_combine(m, s, x, 1.f);
    }
    __shared__ float sM[256], sS[256];
    int t = threadIdx.x;
    sM[t] = m; sS[t] = s;
    __syncthreads();
    for (int off = 128; off; off >>= 1) {
        if (t < off) {
            float mm = sM[t], ssv = sS[t];
            ms_combine(mm, ssv, sM[t + off], sS[t + off]);
            sM[t] = mm; sS[t] = ssv;
        }
        __syncthreads();
    }
    if (t == 0) { part[(h * 64 + chunk) * 2] = sM[0]; part[(h * 64 + chunk) * 2 + 1] = sS[0]; }
}

__global__ __launch_bounds__(256) void k2b_head_final(const float* __restrict__ part,
                                                      float* __restrict__ hmax,
                                                      float* __restrict__ hinv,
                                                      float* __restrict__ z) {
    int t = threadIdx.x, lane = t & 31, wv = t >> 5;
    for (int i = t; i < 2048; i += 256) z[i] = 0.f;   // zero Z for K3
    float m = -3.0e38f, s = 0.f;
    for (int c = lane; c < 64; c += 32)
        ms_combine(m, s, part[(wv * 64 + c) * 2], part[(wv * 64 + c) * 2 + 1]);
#pragma unroll
    for (int off = 16; off; off >>= 1)
        ms_combine(m, s, __shfl_xor(m, off, 32), __shfl_xor(s, off, 32));
    if (lane == 0) { hmax[wv] = m; hinv[wv] = 1.f / s; }
}

// -------------------------------- K3 ---------------------------------------
__global__ __launch_bounds__(128) void k3_values_z(
        const float* __restrict__ node, const __bf16* __restrict__ wvbf,
        const float* __restrict__ bv, const float* __restrict__ logits,
        const float* __restrict__ hmax, const float* __restrict__ hinv,
        float* __restrict__ zGlobal) {
    __shared__ __bf16 sX[64 * XPITCH];      // 33792 B
    __shared__ __bf16 sW[2 * WSLICE];       // 40960 B
    __shared__ float  sWgt[64 * 8];         //  2048 B
    __shared__ float  sZ[2048];             //  8192 B
    const int tid = threadIdx.x;
    const int lane = tid & 31, wv = tid >> 5;
    const int l15 = lane & 15, hi = lane >> 4;
    const int base = blockIdx.x * 64;
    const int m0 = wv * 16;

    stage_slice_async(wvbf, 0, sW, tid);

    const float4* nf4 = (const float4*)node;
    for (int i = tid; i < 64 * 64; i += 128) {
        int r = i >> 6, c4 = i & 63;
        int n = base + r;
        float4 v = make_float4(0.f, 0.f, 0.f, 0.f);
        if (n < NNODES) v = nf4[(size_t)n * 64 + c4];
        __bf16* d = &sX[r * XPITCH + c4 * 4];
        d[0] = (__bf16)v.x; d[1] = (__bf16)v.y; d[2] = (__bf16)v.z; d[3] = (__bf16)v.w;
    }
    for (int i = tid; i < 2048; i += 128) sZ[i] = 0.f;
    for (int i = tid; i < 512; i += 128) {
        int r = i >> 3, h = i & 7;
        int n = base + r;
        float w = 0.f;
        if (n < NNODES) w = __expf(logits[(size_t)h * NNODES + n] - hmax[h]) * hinv[h];
        sWgt[i] = w;
    }

    v8f acc[16];
#pragma unroll
    for (int jt = 0; jt < 16; ++jt)
#pragma unroll
        for (int e = 0; e < 8; ++e) acc[jt][e] = 0.f;

    wait_async_all();
    __syncthreads();

    for (int kt = 0; kt < 8; ++kt) {
        const __bf16* cur = sW + (kt & 1) * WSLICE;
        if (kt < 7) stage_slice_async(wvbf, kt + 1, sW + ((kt + 1) & 1) * WSLICE, tid);

        v16bf A = load_a_frag(sX, m0, lane, kt);
        v16bf Bc = load_b_frag(cur, 0, lane);
#pragma unroll
        for (int jt = 0; jt < 16; ++jt) {
            v16bf Bn = Bc;
            if (jt < 15) Bn = load_b_frag(cur, jt + 1, lane);
            acc[jt] = __builtin_amdgcn_wmma_f32_16x16x32_bf16(
                false, A, false, Bc, (short)0, acc[jt], false, false);
            Bc = Bn;
        }
        wait_async_all();
        __syncthreads();
    }

    // bias + normalize
    float ss[8];
#pragma unroll
    for (int r = 0; r < 8; ++r) ss[r] = 0.f;
#pragma unroll
    for (int jt = 0; jt < 16; ++jt) {
        float bvv = bv[jt * 16 + l15];
#pragma unroll
        for (int r = 0; r < 8; ++r) {
            float t = acc[jt][r] + bvv;
            acc[jt][r] = t;
            ss[r] += t * t;
        }
    }
#pragma unroll
    for (int m = 1; m <= 8; m <<= 1)
#pragma unroll
        for (int r = 0; r < 8; ++r) ss[r] += __shfl_xor(ss[r], m, 32);
#pragma unroll
    for (int r = 0; r < 8; ++r) {
        float sc = 1.f / fmaxf(sqrtf(ss[r]), 1e-12f);
#pragma unroll
        for (int jt = 0; jt < 16; ++jt) acc[jt][r] *= sc;
    }

    // Z[h][c] += sum_rows w[row][h]*values[row][c]
    float wreg[8][8];
#pragma unroll
    for (int r = 0; r < 8; ++r)
#pragma unroll
        for (int h = 0; h < 8; ++h) wreg[r][h] = sWgt[(m0 + hi * 8 + r) * 8 + h];
#pragma unroll
    for (int jt = 0; jt < 16; ++jt) {
        int c = jt * 16 + l15;
#pragma unroll
        for (int h = 0; h < 8; ++h) {
            float s = 0.f;
#pragma unroll
            for (int r = 0; r < 8; ++r) s += wreg[r][h] * acc[jt][r];
            atomicAdd(&sZ[h * 256 + c], s);
        }
    }
    __syncthreads();
    for (int i = tid; i < 2048; i += 128) unsafeAtomicAdd(&zGlobal[i], sZ[i]);
}

// -------------------------------- K4 ---------------------------------------
__global__ __launch_bounds__(256) void k4_attn_u(
        const float* __restrict__ z, const float* __restrict__ in_w,
        const float* __restrict__ in_b, const float* __restrict__ op_w,
        const float* __restrict__ op_b, const float* __restrict__ Wo,
        const float* __restrict__ bo, const float* __restrict__ Wk,
        const float* __restrict__ bk, float* __restrict__ u, float* __restrict__ c0) {
    __shared__ float s1[256], s2[256], red[256];
    int t = threadIdx.x, h = t >> 5;
    float a1 = in_b[512 + t];
    for (int i = 0; i < 256; ++i) a1 += z[h * 256 + i] * in_w[(size_t)(512 + t) * 256 + i];
    s1[t] = a1;
    __syncthreads();
    float a2 = op_b[t];
    for (int j = 0; j < 256; ++j) a2 += op_w[(size_t)t * 256 + j] * s1[j];
    s2[t] = a2;
    __syncthreads();
    float a3 = bo[t];
    for (int j = 0; j < 256; ++j) a3 += Wo[(size_t)t * 256 + j] * s2[j];
    red[t] = a3 * a3;
    __syncthreads();
    for (int off = 128; off; off >>= 1) {
        if (t < off) red[t] += red[t + off];
        __syncthreads();
    }
    float a = a3 / fmaxf(sqrtf(red[0]), 1e-12f);
    __syncthreads();
    s1[t] = a;
    __syncthreads();
    float uu = 0.f;
    for (int j = 0; j < 256; ++j) uu += Wk[(size_t)j * 256 + t] * s1[j];
    u[t] = uu;
    red[t] = bk[t] * s1[t];
    __syncthreads();
    for (int off = 128; off; off >>= 1) {
        if (t < off) red[t] += red[t + off];
        __syncthreads();
    }
    if (t == 0) c0[0] = red[0];
}

// -------------------------------- K5 ---------------------------------------
__global__ __launch_bounds__(256) void k5_scores(
        const float* __restrict__ node, const unsigned char* __restrict__ mask,
        const float* __restrict__ u, const float* __restrict__ c0,
        const float* __restrict__ scale, float* __restrict__ scores) {
    __shared__ float sU[256];
    int tid = threadIdx.x, lane = tid & 31, wv = tid >> 5;
    sU[tid] = u[tid];
    __syncthreads();
    int n = blockIdx.x * 8 + wv;
    if (n >= NNODES) return;
    const float4* xr = (const float4*)(node + (size_t)n * 256);
    float p = 0.f;
#pragma unroll
    for (int q = 0; q < 2; ++q) {
        float4 xv = xr[lane * 2 + q];
        int cb = lane * 8 + q * 4;
        p += xv.x * sU[cb] + xv.y * sU[cb + 1] + xv.z * sU[cb + 2] + xv.w * sU[cb + 3];
    }
#pragma unroll
    for (int off = 16; off; off >>= 1) p += __shfl_xor(p, off, 32);
    if (lane == 0)
        scores[n] = mask[n] ? NEGV : (p + c0[0]) * scale[n];
}

// -------------------------------- K6/K7 ------------------------------------
__global__ __launch_bounds__(256) void k6a_score_partials(const float* __restrict__ scores,
                                                          float* __restrict__ part) {
    int chunk = blockIdx.x;
    int start = chunk * 3125, end = min(start + 3125, NNODES);
    float m = -3.0e38f, s = 0.f;
    for (int i = start + threadIdx.x; i < end; i += 256) ms_combine(m, s, scores[i], 1.f);
    __shared__ float sM[256], sS[256];
    int t = threadIdx.x;
    sM[t] = m; sS[t] = s;
    __syncthreads();
    for (int off = 128; off; off >>= 1) {
        if (t < off) {
            float mm = sM[t], ssv = sS[t];
            ms_combine(mm, ssv, sM[t + off], sS[t + off]);
            sM[t] = mm; sS[t] = ssv;
        }
        __syncthreads();
    }
    if (t == 0) { part[chunk * 2] = sM[0]; part[chunk * 2 + 1] = sS[0]; }
}

__global__ __launch_bounds__(32) void k6b_score_final(const float* __restrict__ part,
                                                      float* __restrict__ smax,
                                                      float* __restrict__ sinv) {
    int lane = threadIdx.x;
    float m = -3.0e38f, s = 0.f;
    for (int c = lane; c < 64; c += 32) ms_combine(m, s, part[c * 2], part[c * 2 + 1]);
#pragma unroll
    for (int off = 16; off; off >>= 1)
        ms_combine(m, s, __shfl_xor(m, off, 32), __shfl_xor(s, off, 32));
    if (lane == 0) { smax[0] = m; sinv[0] = 1.f / s; }
}

__global__ __launch_bounds__(256) void k7_probs(const float* __restrict__ scores,
                                                const float* __restrict__ smax,
                                                const float* __restrict__ sinv,
                                                float* __restrict__ out) {
    int n = blockIdx.x * 256 + threadIdx.x;
    if (n < NNODES) out[n] = __expf(scores[n] - smax[0]) * sinv[0];
}

// ---------------------------------------------------------------------------
extern "C" void kernel_launch(void* const* d_in, const int* in_sizes, int n_in,
                              void* d_out, int out_size, void* d_ws, size_t ws_size,
                              hipStream_t stream) {
    (void)in_sizes; (void)n_in; (void)out_size; (void)ws_size;
    const float*         ctx   = (const float*)d_in[0];
    const float*         node  = (const float*)d_in[1];
    const unsigned char* mask  = (const unsigned char*)d_in[2];
    const float*         Wc    = (const float*)d_in[3];
    const float*         bc    = (const float*)d_in[4];
    const float*         Wk    = (const float*)d_in[5];
    const float*         bk    = (const float*)d_in[6];
    const float*         Wv    = (const float*)d_in[7];
    const float*         bv    = (const float*)d_in[8];
    const float*         in_w  = (const float*)d_in[9];
    const float*         in_b  = (const float*)d_in[10];
    const float*         op_w  = (const float*)d_in[11];
    const float*         op_b  = (const float*)d_in[12];
    const float*         Wo    = (const float*)d_in[13];
    const float*         bo    = (const float*)d_in[14];
    float* out = (float*)d_out;

    char* wsb = (char*)d_ws;
    __bf16* wkbf   = (__bf16*)(wsb + OFF_WK);
    __bf16* wvbf   = (__bf16*)(wsb + OFF_WV);
    float*  G      = (float*)(wsb + OFF_G);
    float*  scl    = (float*)(wsb + OFF_SCALE);
    float*  logits = (float*)(wsb + OFF_LOGITS);
    float*  scores = (float*)(wsb + OFF_SCORES);
    float*  part   = (float*)(wsb + OFF_PART);
    float*  hmax   = (float*)(wsb + OFF_HMAX);
    float*  hinv   = (float*)(wsb + OFF_HINV);
    float*  zg     = (float*)(wsb + OFF_Z);
    float*  u      = (float*)(wsb + OFF_U);
    float*  c0     = (float*)(wsb + OFF_C0);
    float*  spart  = (float*)(wsb + OFF_SPART);
    float*  smax   = (float*)(wsb + OFF_SMAX);
    float*  sinv   = (float*)(wsb + OFF_SINV);

    kw_conv<<<256, 256, 0, stream>>>(Wk, Wv, wkbf, wvbf);
    k0_query<<<1, 256, 0, stream>>>(ctx, Wc, bc, in_w, in_b, G);
    k1_keys_logits<<<3125, 128, 0, stream>>>(node, mask, wkbf, bk, G, scl, logits);
    k2a_head_partials<<<512, 256, 0, stream>>>(logits, part);
    k2b_head_final<<<1, 256, 0, stream>>>(part, hmax, hinv, zg);
    k3_values_z<<<3125, 128, 0, stream>>>(node, wvbf, bv, logits, hmax, hinv, zg);
    k4_attn_u<<<1, 256, 0, stream>>>(zg, in_w, in_b, op_w, op_b, Wo, bo, Wk, bk, u, c0);
    k5_scores<<<25000, 256, 0, stream>>>(node, mask, u, c0, scl, scores);
    k6a_score_partials<<<64, 256, 0, stream>>>(scores, spart);
    k6b_score_final<<<1, 32, 0, stream>>>(spart, smax, sinv);
    k7_probs<<<782, 256, 0, stream>>>(scores, smax, sinv, out);
}
